// BitNetTransformerLayer_24670292148854
// MI455X (gfx1250) — 16-bit and 8-bit matrix operand order repaired and checked by reading, compiled, not run on hardware
//
#include <hip/hip_runtime.h>
#include <stdint.h>


constexpr int BB = 2, TT = 2048, DD = 2048, FF = 8192, HH = 16;
constexpr int MM = BB * TT;
#define DEVFN __device__ __forceinline__

typedef int      v8i  __attribute__((ext_vector_type(8)));
typedef float    v8f  __attribute__((ext_vector_type(8)));
typedef _Float16 v16h __attribute__((ext_vector_type(16)));
typedef int      v4i  __attribute__((ext_vector_type(4)));

union FragI { v8i v; int d[8]; };
union FragH { v16h v; unsigned u[8]; _Float16 e[16]; };

#if defined(__has_builtin)
# if __has_builtin(__builtin_amdgcn_global_load_async_to_lds_b128)
#  define HAVE_ASYNC_LDS 1
# endif
#endif
#ifndef HAVE_ASYNC_LDS
# define HAVE_ASYNC_LDS 0
#endif

#if HAVE_ASYNC_LDS
typedef __attribute__((address_space(1))) v4i gv4i;
typedef __attribute__((address_space(3))) v4i lv4i;
# if __has_builtin(__builtin_amdgcn_s_wait_asynccnt)
#  define WAIT_ASYNC(n) __builtin_amdgcn_s_wait_asynccnt(n)
# else
#  define WAIT_ASYNC(n) asm volatile("s_wait_asynccnt %0" :: "n"(n) : "memory")
# endif
#else
# define WAIT_ASYNC(n) ((void)0)
#endif

DEVFN float block_reduce_sum(float v, float* red) {
  int t = threadIdx.x;
  __syncthreads();
  red[t] = v; __syncthreads();
  for (int s = 128; s > 0; s >>= 1) { if (t < s) red[t] += red[t + s]; __syncthreads(); }
  return red[0];
}
DEVFN float block_reduce_max(float v, float* red) {
  int t = threadIdx.x;
  __syncthreads();
  red[t] = v; __syncthreads();
  for (int s = 128; s > 0; s >>= 1) { if (t < s) red[t] = fmaxf(red[t], red[t + s]); __syncthreads(); }
  return red[0];
}

__global__ __launch_bounds__(256)
void abs_partial_kernel(const float* __restrict__ W, size_t n, float* __restrict__ part) {
  __shared__ float red[256];
  float s = 0.f;
  for (size_t i = (size_t)blockIdx.x * 256 + threadIdx.x; i < n; i += (size_t)gridDim.x * 256)
    s += fabsf(W[i]);
  s = block_reduce_sum(s, red);
  if (threadIdx.x == 0) part[blockIdx.x] = s;
}

__global__ __launch_bounds__(256)
void scale_finalize_kernel(const float* __restrict__ part, int nb, float n, float* __restrict__ out) {
  __shared__ float red[256];
  float s = 0.f;
  for (int i = threadIdx.x; i < nb; i += 256) s += part[i];
  s = block_reduce_sum(s, red);
  if (threadIdx.x == 0) out[0] = s / n;
}

__global__ __launch_bounds__(256)
void quant_w_kernel(const float* __restrict__ W, signed char* __restrict__ Q,
                    const float* __restrict__ sc, size_t n) {
  const float inv = 1.f / (sc[0] + 1e-8f);
  for (size_t i = (size_t)blockIdx.x * 256 + threadIdx.x; i < n; i += (size_t)gridDim.x * 256) {
    float q = rintf(W[i] * inv);
    q = fminf(fmaxf(q, -1.f), 1.f);
    Q[i] = (signed char)q;
  }
}

template <int COLS>
__global__ __launch_bounds__(256)
void ln_quant_kernel(const float* __restrict__ X, const float* __restrict__ G,
                     const float* __restrict__ Bv, signed char* __restrict__ Q,
                     float* __restrict__ SX) {
  constexpr int E = COLS / 256;
  __shared__ float red[256];
  const int row = blockIdx.x, t = threadIdx.x;
  const float* x = X + (size_t)row * COLS;
  float v[E];
  float s = 0.f;
#pragma unroll
  for (int i = 0; i < E; ++i) { v[i] = x[t + i * 256]; s += v[i]; }
  const float mu = block_reduce_sum(s, red) * (1.f / COLS);
  float q = 0.f;
#pragma unroll
  for (int i = 0; i < E; ++i) { float d = v[i] - mu; q += d * d; }
  const float var  = block_reduce_sum(q, red) * (1.f / COLS);
  const float rstd = rsqrtf(var + 1e-5f);
  float mx = 0.f;
#pragma unroll
  for (int i = 0; i < E; ++i) {
    v[i] = (v[i] - mu) * rstd * G[t + i * 256] + Bv[t + i * 256];
    mx = fmaxf(mx, fabsf(v[i]));
  }
  const float gamma = block_reduce_max(mx, red);
  const float inv = 127.f / (gamma + 1e-8f);
#pragma unroll
  for (int i = 0; i < E; ++i) {
    float qv = fminf(fmaxf(rintf(v[i] * inv), -128.f), 127.f);
    Q[(size_t)row * COLS + t + i * 256] = (signed char)qv;
  }
  if (t == 0) SX[row] = gamma * (1.f / 127.f);
}

__global__ __launch_bounds__(256)
void glu_ln_quant_kernel(const _Float16* __restrict__ Gt, const _Float16* __restrict__ Ut,
                         const float* __restrict__ G, const float* __restrict__ Bv,
                         signed char* __restrict__ Q, float* __restrict__ SX) {
  constexpr int COLS = FF, E = COLS / 256;
  __shared__ float red[256];
  const int row = blockIdx.x, t = threadIdx.x;
  const size_t base = (size_t)row * COLS;
  float v[E];
  float s = 0.f;
#pragma unroll
  for (int i = 0; i < E; ++i) {
    float gv = (float)Gt[base + t + i * 256];
    float uv = (float)Ut[base + t + i * 256];
    v[i] = uv / (1.f + expf(-gv));
    s += v[i];
  }
  const float mu = block_reduce_sum(s, red) * (1.f / COLS);
  float q = 0.f;
#pragma unroll
  for (int i = 0; i < E; ++i) { float d = v[i] - mu; q += d * d; }
  const float var  = block_reduce_sum(q, red) * (1.f / COLS);
  const float rstd = rsqrtf(var + 1e-5f);
  float mx = 0.f;
#pragma unroll
  for (int i = 0; i < E; ++i) {
    v[i] = (v[i] - mu) * rstd * G[t + i * 256] + Bv[t + i * 256];
    mx = fmaxf(mx, fabsf(v[i]));
  }
  const float gamma = block_reduce_max(mx, red);
  const float inv = 127.f / (gamma + 1e-8f);
#pragma unroll
  for (int i = 0; i < E; ++i) {
    float qv = fminf(fmaxf(rintf(v[i] * inv), -128.f), 127.f);
    Q[base + t + i * 256] = (signed char)qv;
  }
  if (t == 0) SX[row] = gamma * (1.f / 127.f);
}

DEVFN int koffA8(int v, int hi) { return ((v >> 1) << 4) + hi * 8  + ((v & 1) << 2); }
DEVFN int koffB8(int v, int hi) { return ((v >> 2) << 5) + hi * 16 + ((v & 3) << 2); }
DEVFN int lds_swz(int row, int koff) {
  return row * 64 + ((((koff >> 4) ^ (row >> 2)) & 3) << 4) + (koff & 15);
}

DEVFN void stage_slab(const signed char* __restrict__ gsrc, signed char* lbuf,
                      int ldK, int k0, int t) {
#pragma unroll
  for (int c = t; c < 512; c += 256) {
    const int row = c >> 2, seg = c & 3;
    const signed char* gp = gsrc + (size_t)row * ldK + k0 + seg * 16;
    signed char* lp = lbuf + lds_swz(row, seg * 16);
#if HAVE_ASYNC_LDS
    __builtin_amdgcn_global_load_async_to_lds_b128(
        (gv4i*)(void*)gp, (lv4i*)(void*)lp, 0, 0);
#else
    *(int4*)lp = *(const int4*)gp;
#endif
  }
}

__global__ __launch_bounds__(256)
void gemm_iu8_kernel(const signed char* __restrict__ A, const signed char* __restrict__ W,
                     const float* __restrict__ sx, const float* __restrict__ swp,
                     const float* __restrict__ resid,
                     float* __restrict__ outF, _Float16* __restrict__ outH,
                     int M, int N, int K, int mode) {
  __shared__ signed char smA[2][128 * 64];
  __shared__ signed char smB[2][128 * 64];
  const int t = threadIdx.x;
  const int lane = t & 31, wave = t >> 5;
  const int lo = lane & 15, hi = lane >> 4;
  const int wm = wave >> 2, wn = wave & 3;
  const int bm0 = blockIdx.y * 128, bn0 = blockIdx.x * 128;
  const signed char* Ab = A + (size_t)bm0 * K;
  const signed char* Wb = W + (size_t)bn0 * K;
  const float sw = swp[0];
  const int nk = K >> 6;

  stage_slab(Ab, smA[0], K, 0, t);
  stage_slab(Wb, smB[0], K, 0, t);

  v8i acc[4][2] = {};
  for (int kb = 0; kb < nk; ++kb) {
    const int cur = kb & 1;
    if (kb + 1 < nk) {
      stage_slab(Ab, smA[cur ^ 1], K, (kb + 1) << 6, t);
      stage_slab(Wb, smB[cur ^ 1], K, (kb + 1) << 6, t);
      WAIT_ASYNC(4);
    } else {
      WAIT_ASYNC(0);
    }
    __syncthreads();

    const signed char* bufA = smA[cur];
    const signed char* bufB = smB[cur];
    FragI a[4], b[2];
#pragma unroll
    for (int mi = 0; mi < 4; ++mi) {
      const int row = wm * 64 + mi * 16 + lo;
#pragma unroll
      for (int v = 0; v < 8; ++v)
        a[mi].d[v] = *(const int*)(bufA + lds_swz(row, koffA8(v, hi)));
    }
#pragma unroll
    for (int ni = 0; ni < 2; ++ni) {
      const int col = wn * 32 + ni * 16 + lo;
#pragma unroll
      for (int v = 0; v < 8; ++v)
        b[ni].d[v] = *(const int*)(bufB + lds_swz(col, koffA8(v, hi)));
    }
#pragma unroll
    for (int mi = 0; mi < 4; ++mi)
#pragma unroll
      for (int ni = 0; ni < 2; ++ni)
        acc[mi][ni] = __builtin_amdgcn_wmma_i32_16x16x64_iu8(
             true, a[mi].v,  true, b[ni].v,
            acc[mi][ni],  false,  false);

    __syncthreads();
  }

#pragma unroll
  for (int mi = 0; mi < 4; ++mi)
#pragma unroll
    for (int ni = 0; ni < 2; ++ni) {
      const int n = bn0 + wn * 32 + ni * 16 + lo;
#pragma unroll
      for (int v = 0; v < 8; ++v) {
        const int m = bm0 + wm * 64 + mi * 16 + 8 * hi + v;
        float val = (float)acc[mi][ni][v] * sx[m] * sw;
        const size_t idx = (size_t)m * N + n;
        if (mode == 2)       outH[idx] = (_Float16)val;
        else if (mode == 1)  outF[idx] = val + resid[idx];
        else                 outF[idx] = val;
      }
    }
}

DEVFN int koffA16(int v, int hi) { return ((v >> 2) << 4) + hi * 8 + ((v & 3) << 1); }

__global__ __launch_bounds__(256)
void attn_scores_kernel(const _Float16* __restrict__ Q16, const _Float16* __restrict__ K16,
                        float* __restrict__ S) {
  const int bh = blockIdx.z, b = bh >> 4, h = bh & 15;
  const int lane = threadIdx.x & 31, wave = threadIdx.x >> 5;
  const int lo = lane & 15, hi = lane >> 4;
  const int wm = wave >> 1, wn = wave & 1;
  const int bm = blockIdx.y * 128 + wm * 32;
  const int bn = blockIdx.x * 64  + wn * 32;
  const _Float16* Qb = Q16 + (size_t)b * TT * DD + h * 128;
  const _Float16* Kb = K16 + (size_t)b * TT * DD + h * 128;

  v8f acc[2][2] = {};
  for (int k0 = 0; k0 < 128; k0 += 32) {
    FragH a[2], bf[2];
#pragma unroll
    for (int mi = 0; mi < 2; ++mi) {
      const _Float16* qp = Qb + (size_t)(bm + mi * 16 + lo) * DD + k0;
#pragma unroll
      for (int v = 0; v < 8; ++v) a[mi].u[v] = *(const unsigned*)(qp + koffA16(v, hi));
    }
#pragma unroll
    for (int ni = 0; ni < 2; ++ni) {
      const _Float16* kp = Kb + (size_t)(bn + ni * 16 + lo) * DD + k0;
#pragma unroll
      for (int v = 0; v < 8; ++v) bf[ni].u[v] = *(const unsigned*)(kp + koffA16(v, hi));
    }
#pragma unroll
    for (int mi = 0; mi < 2; ++mi)
#pragma unroll
      for (int ni = 0; ni < 2; ++ni)
        acc[mi][ni] = __builtin_amdgcn_wmma_f32_16x16x32_f16(
            false, a[mi].v, false, bf[ni].v, (short)0, acc[mi][ni], false, false);
  }
  const float rs = 0.08838834764831845f;
#pragma unroll
  for (int mi = 0; mi < 2; ++mi)
#pragma unroll
    for (int ni = 0; ni < 2; ++ni) {
      const int n = bn + ni * 16 + lo;
#pragma unroll
      for (int v = 0; v < 8; ++v) {
        const int m = bm + mi * 16 + 8 * hi + v;
        S[((size_t)bh * TT + m) * TT + n] = acc[mi][ni][v] * rs;
      }
    }
}

__global__ __launch_bounds__(256)
void softmax_kernel(float* __restrict__ S) {
  __shared__ float red[256];
  float* row = S + (size_t)blockIdx.x * TT;
  const int t = threadIdx.x;
  float v[TT / 256];
  float mx = -3.4e38f;
#pragma unroll
  for (int i = 0; i < TT / 256; ++i) { v[i] = row[t + i * 256]; mx = fmaxf(mx, v[i]); }
  mx = block_reduce_max(mx, red);
  float s = 0.f;
#pragma unroll
  for (int i = 0; i < TT / 256; ++i) { v[i] = expf(v[i] - mx); s += v[i]; }
  s = block_reduce_sum(s, red);
  const float inv = 1.f / s;
#pragma unroll
  for (int i = 0; i < TT / 256; ++i) row[t + i * 256] = v[i] * inv;
}

__global__ __launch_bounds__(256)
void attn_pv_kernel(const float* __restrict__ S, const _Float16* __restrict__ V16,
                    float* __restrict__ AO) {
  const int bh = blockIdx.z, b = bh >> 4, h = bh & 15;
  const int lane = threadIdx.x & 31, wave = threadIdx.x >> 5;
  const int lo = lane & 15, hi = lane >> 4;
  const int wm = wave >> 1, wn = wave & 1;
  const int bm = blockIdx.y * 128 + wm * 32;
  const int bn = blockIdx.x * 64  + wn * 32;
  const float*    Sb = S   + (size_t)bh * TT * TT;
  const _Float16* Vb = V16 + (size_t)b * TT * DD + h * 128;

  v8f acc[2][2] = {};
  for (int k0 = 0; k0 < TT; k0 += 32) {
    FragH a[2], bf[2];
#pragma unroll
    for (int mi = 0; mi < 2; ++mi) {
      const float* sp = Sb + (size_t)(bm + mi * 16 + lo) * TT + k0;
#pragma unroll
      for (int v = 0; v < 8; ++v) {
        float2 f2 = *(const float2*)(sp + koffA16(v, hi));
        a[mi].e[2 * v]     = (_Float16)f2.x;
        a[mi].e[2 * v + 1] = (_Float16)f2.y;
      }
    }
#pragma unroll
    for (int ni = 0; ni < 2; ++ni) {
      const int n = bn + ni * 16 + lo;
#pragma unroll
      for (int v = 0; v < 8; ++v) {
        const int k = koffA16(v, hi);
        bf[ni].e[2 * v]     = Vb[(size_t)(k0 + k)     * DD + n];
        bf[ni].e[2 * v + 1] = Vb[(size_t)(k0 + k + 1) * DD + n];
      }
    }
#pragma unroll
    for (int mi = 0; mi < 2; ++mi)
#pragma unroll
      for (int ni = 0; ni < 2; ++ni)
        acc[mi][ni] = __builtin_amdgcn_wmma_f32_16x16x32_f16(
            false, a[mi].v, false, bf[ni].v, (short)0, acc[mi][ni], false, false);
  }
#pragma unroll
  for (int mi = 0; mi < 2; ++mi)
#pragma unroll
    for (int ni = 0; ni < 2; ++ni) {
      const int n = bn + ni * 16 + lo;
#pragma unroll
      for (int v = 0; v < 8; ++v) {
        const int m = bm + mi * 16 + 8 * hi + v;
        AO[(size_t)(b * TT + m) * DD + h * 128 + n] = acc[mi][ni][v];
      }
    }
}

extern "C" void kernel_launch(void* const* d_in, const int* in_sizes, int n_in,
                              void* d_out, int out_size, void* d_ws, size_t ws_size,
                              hipStream_t stream) {
  (void)in_sizes; (void)n_in; (void)out_size; (void)ws_size;
  const float* x    = (const float*)d_in[0];
  const float* wsrc[7] = {(const float*)d_in[1], (const float*)d_in[2], (const float*)d_in[3],
                          (const float*)d_in[4], (const float*)d_in[5], (const float*)d_in[6],
                          (const float*)d_in[7]};
  const float* ln1g = (const float*)d_in[8];
  const float* ln1b = (const float*)d_in[9];
  const float* ln2g = (const float*)d_in[10];
  const float* ln2b = (const float*)d_in[11];
  const float* lnag = (const float*)d_in[12];
  const float* lnab = (const float*)d_in[13];
  const float* lnfg = (const float*)d_in[14];
  const float* lnfb = (const float*)d_in[15];

  char* ws = (char*)d_ws;
  size_t off = 0;
  auto alloc = [&](size_t bytes) -> void* {
    off = (off + 255) & ~(size_t)255;
    void* p = ws + off; off += bytes; return p;
  };

  float* scales = (float*)alloc(8 * sizeof(float));
  float* part   = (float*)alloc(1024 * sizeof(float));
  const size_t wn[7] = {(size_t)DD * DD, (size_t)DD * DD, (size_t)DD * DD, (size_t)DD * DD,
                        (size_t)FF * DD, (size_t)FF * DD, (size_t)DD * FF};
  signed char* w8[7];
  for (int i = 0; i < 7; ++i) w8[i] = (signed char*)alloc(wn[i]);

  signed char* xq1 = (signed char*)alloc((size_t)MM * DD);
  float*       sx1 = (float*)alloc((size_t)MM * 4);
  _Float16*    q16 = (_Float16*)alloc((size_t)MM * DD * 2);
  _Float16*    k16 = (_Float16*)alloc((size_t)MM * DD * 2);
  _Float16*    v16 = (_Float16*)alloc((size_t)MM * DD * 2);
  float*       ao  = (float*)alloc((size_t)MM * DD * 4);
  signed char* aoq = (signed char*)alloc((size_t)MM * DD);
  float*       sxa = (float*)alloc((size_t)MM * 4);
  float*       x1  = (float*)alloc((size_t)MM * DD * 4);
  signed char* xq2 = (signed char*)alloc((size_t)MM * DD);
  float*       sx2 = (float*)alloc((size_t)MM * 4);
  _Float16*    g16 = (_Float16*)alloc((size_t)MM * FF * 2);
  _Float16*    u16 = (_Float16*)alloc((size_t)MM * FF * 2);
  signed char* hq  = (signed char*)alloc((size_t)MM * FF);
  float*       sxh = (float*)alloc((size_t)MM * 4);

  float* outx = (float*)d_out;
  float* attn = outx + (size_t)MM * DD;

  for (int i = 0; i < 7; ++i) {
    abs_partial_kernel<<<1024, 256, 0, stream>>>(wsrc[i], wn[i], part);
    scale_finalize_kernel<<<1, 256, 0, stream>>>(part, 1024, (float)wn[i], scales + i);
    quant_w_kernel<<<4096, 256, 0, stream>>>(wsrc[i], w8[i], scales + i, wn[i]);
  }

  ln_quant_kernel<DD><<<MM, 256, 0, stream>>>(x, ln1g, ln1b, xq1, sx1);

  dim3 gD(DD / 128, MM / 128);
  gemm_iu8_kernel<<<gD, 256, 0, stream>>>(xq1, w8[0], sx1, scales + 0, nullptr, nullptr, q16, MM, DD, DD, 2);
  gemm_iu8_kernel<<<gD, 256, 0, stream>>>(xq1, w8[1], sx1, scales + 1, nullptr, nullptr, k16, MM, DD, DD, 2);
  gemm_iu8_kernel<<<gD, 256, 0, stream>>>(xq1, w8[2], sx1, scales + 2, nullptr, nullptr, v16, MM, DD, DD, 2);

  attn_scores_kernel<<<dim3(TT / 64, TT / 128, BB * HH), 256, 0, stream>>>(q16, k16, attn);
  softmax_kernel<<<BB * HH * TT, 256, 0, stream>>>(attn);
  attn_pv_kernel<<<dim3(128 / 64, TT / 128, BB * HH), 256, 0, stream>>>(attn, v16, ao);

  ln_quant_kernel<DD><<<MM, 256, 0, stream>>>(ao, lnag, lnab, aoq, sxa);
  gemm_iu8_kernel<<<gD, 256, 0, stream>>>(aoq, w8[3], sxa, scales + 3, x, x1, nullptr, MM, DD, DD, 1);

  ln_quant_kernel<DD><<<MM, 256, 0, stream>>>(x1, ln2g, ln2b, xq2, sx2);
  dim3 gF(FF / 128, MM / 128);
  gemm_iu8_kernel<<<gF, 256, 0, stream>>>(xq2, w8[4], sx2, scales + 4, nullptr, nullptr, g16, MM, FF, DD, 2);
  gemm_iu8_kernel<<<gF, 256, 0, stream>>>(xq2, w8[5], sx2, scales + 5, nullptr, nullptr, u16, MM, FF, DD, 2);

  glu_ln_quant_kernel<<<MM, 256, 0, stream>>>(g16, u16, lnfg, lnfb, hq, sxh);
  gemm_iu8_kernel<<<gD, 256, 0, stream>>>(hq, w8[6], sxh, scales + 6, x1, outx, nullptr, MM, DD, FF, 1);
}
